// rmc2_criteo_70935679861559
// MI455X (gfx1250) — matrix operand order repaired and checked by reading, not compiled, not run on hardware
//
#include <hip/hip_runtime.h>
#include <hip/hip_bf16.h>
#include <math.h>
#include <stdint.h>

typedef __attribute__((ext_vector_type(16))) _Float16 v16h;
typedef __attribute__((ext_vector_type(8)))  float    v8f;

// ---------------------------------------------------------------------------
// CDNA5 async global->LDS copies (ASYNCcnt-tracked, no VGPR round trip).
// Per ISA 10.2 aperture rules, addr[31:0] of a generic shared pointer is the
// LDS byte address, which is what VDST of the async instruction expects.
// ---------------------------------------------------------------------------
__device__ inline unsigned lds_off(const void* p) {
  return (unsigned)(uintptr_t)p;
}
__device__ inline void async_ld_b128(unsigned lds, const void* g) {
  asm volatile("global_load_async_to_lds_b128 %0, %1, off"
               :: "v"(lds), "v"(g) : "memory");
}
__device__ inline void async_ld_b64(unsigned lds, const void* g) {
  asm volatile("global_load_async_to_lds_b64 %0, %1, off"
               :: "v"(lds), "v"(g) : "memory");
}
__device__ inline void s_wait_async0() {
  asm volatile("s_wait_asynccnt 0x0" ::: "memory");
}

// ---------------------------------------------------------------------------
// Fragment loaders (wave32); this file's own contiguous K order, same for A and B.
// A 16x32 f16: lanes 0-15 row M=lane, K=0..15; lanes 16-31 K=16..31.
// v16h elem i -> VGPR i/2.
// ---------------------------------------------------------------------------
__device__ inline v16h load_a_frag(const _Float16* base, int stride) {
  const int lane = threadIdx.x & 31;
  const int m    = lane & 15;
  const int kh   = (lane & 16) ? 16 : 0;
  const _Float16* p = base + m * stride + kh;
  v16h a;
#pragma unroll
  for (int r = 0; r < 8; ++r) a[r] = p[r];
#pragma unroll
  for (int r = 0; r < 8; ++r) a[8 + r] = p[8 + r];
  return a;
}

// B 32x16 f16 from K-major [K][N] LDS tile: lanes 0-15 col N=lane, K=0..15;
// lanes 16-31 K=16..31.
__device__ inline v16h load_b_frag(const _Float16* base, int stride) {
  const int lane = threadIdx.x & 31;
  const int n    = lane & 15;
  const int kb   = (lane & 16) ? 16 : 0;
  v16h b;
#pragma unroll
  for (int r = 0; r < 16; ++r) b[r] = base[(kb + r) * stride + n];
  return b;
}

// B 32x16 where B[k][n] = T[n][k]: contiguous 32B per lane (free transpose).
__device__ inline v16h load_bT_frag(const _Float16* base, int stride) {
  const int lane = threadIdx.x & 31;
  const int n    = lane & 15;
  const int kb   = (lane & 16) ? 16 : 0;
  const _Float16* p = base + n * stride + kb;
  v16h b;
#pragma unroll
  for (int r = 0; r < 16; ++r) b[r] = p[r];
  return b;
}

// ---------------------------------------------------------------------------
// WMMA GEMM with double-buffered async LDS staging.
// C[M x N] = act(A[M x K] @ W[K x N]), f16 row-major, M = gridDim.y*64,
// N % 64 == 0, K % 32 == 0. Block = 128 threads (4 waves).
// ---------------------------------------------------------------------------
template <bool RELU>
__global__ __launch_bounds__(128) void gemm_f16(const _Float16* __restrict__ A,
                                                const _Float16* __restrict__ W,
                                                _Float16* __restrict__ C,
                                                int N, int K) {
  __shared__ _Float16 As[2][64][40];  // 64x32 tiles, padded stride
  __shared__ _Float16 Ws[2][32][72];  // 32x64 tiles, padded stride
  const int row0 = blockIdx.y * 64;
  const int col0 = blockIdx.x * 64;
  const int tid  = threadIdx.x;
  const int w    = tid >> 5;
  const int lane = tid & 31;
  const int ra = tid >> 1, pa = tid & 1;  // A staging: 2 thr/row, 16 halfs each
  const int rw = tid >> 2, pw = tid & 3;  // W staging: 4 thr/row, 16 halfs each

  v8f acc[4] = {{}, {}, {}, {}};

  // Prologue: stage K-slab 0 into buffer 0.
  async_ld_b128(lds_off(&As[0][ra][pa * 16]),     A + (size_t)(row0 + ra) * K + pa * 16);
  async_ld_b128(lds_off(&As[0][ra][pa * 16 + 8]), A + (size_t)(row0 + ra) * K + pa * 16 + 8);
  async_ld_b128(lds_off(&Ws[0][rw][pw * 16]),     W + (size_t)rw * N + col0 + pw * 16);
  async_ld_b128(lds_off(&Ws[0][rw][pw * 16 + 8]), W + (size_t)rw * N + col0 + pw * 16 + 8);

  int buf = 0;
  for (int k0 = 0; k0 < K; k0 += 32) {
    s_wait_async0();   // this wave's slab (and any in-flight next-slab) landed
    __syncthreads();   // every wave's slab landed
    if (k0 + 32 < K) { // overlap: DMA next slab while we do math on this one
      const int nb = buf ^ 1;
      const int k1 = k0 + 32;
      async_ld_b128(lds_off(&As[nb][ra][pa * 16]),     A + (size_t)(row0 + ra) * K + k1 + pa * 16);
      async_ld_b128(lds_off(&As[nb][ra][pa * 16 + 8]), A + (size_t)(row0 + ra) * K + k1 + pa * 16 + 8);
      async_ld_b128(lds_off(&Ws[nb][rw][pw * 16]),     W + (size_t)(k1 + rw) * N + col0 + pw * 16);
      async_ld_b128(lds_off(&Ws[nb][rw][pw * 16 + 8]), W + (size_t)(k1 + rw) * N + col0 + pw * 16 + 8);
    }
    const v16h a = load_a_frag(&As[buf][w * 16][0], 40);
#pragma unroll
    for (int c = 0; c < 4; ++c) {
      const v16h b = load_b_frag(&Ws[buf][0][c * 16], 72);
      acc[c] = __builtin_amdgcn_wmma_f32_16x16x32_f16(
          false, a, false, b, (short)0, acc[c], false, false);
    }
    buf ^= 1;
  }

  // C/D layout: VGPR r -> M = r (+8 for lanes 16-31), N = lane&15
  const int hi = (lane & 16) ? 8 : 0;
  const int cn = lane & 15;
#pragma unroll
  for (int c = 0; c < 4; ++c) {
#pragma unroll
    for (int r = 0; r < 8; ++r) {
      float v = acc[c][r];
      if (RELU) v = v > 0.f ? v : 0.f;
      C[(size_t)(row0 + w * 16 + hi + r) * N + (col0 + c * 16 + cn)] = (_Float16)v;
    }
  }
}

// ---------------------------------------------------------------------------
// Interaction: one wave per sample. T = [y1; emb rows] (27x128 padded to
// 32x128 in LDS, gathered via async b64 DMA). Z = T*T^T lower triangle via
// 3 WMMA tiles. Emit R row: [ y1(128) | Zflat(351) | zeros(33) ], width 512.
// ---------------------------------------------------------------------------
__global__ __launch_bounds__(128) void interact_kernel(
    const _Float16* __restrict__ y1, const _Float16* __restrict__ emb,
    const long long* __restrict__ sp, _Float16* __restrict__ R) {
  __shared__ _Float16 T[4][32][136];
  __shared__ float    Z[4][32][33];
  const int w    = threadIdx.x >> 5;
  const int lane = threadIdx.x & 31;
  const int s    = blockIdx.x * 4 + w;

  // Gather T rows via async DMA: row 0 = y1[s], rows 1..26 = emb[idx].
  async_ld_b64(lds_off(&T[w][0][lane * 4]), y1 + (size_t)s * 128 + lane * 4);
#pragma unroll 1
  for (int row = 1; row <= 26; ++row) {
    const long long e = sp[(size_t)s * 26 + (row - 1)];
    async_ld_b64(lds_off(&T[w][row][lane * 4]), emb + (size_t)e * 128 + lane * 4);
  }
#pragma unroll
  for (int row = 27; row < 32; ++row)  // zero padding rows
    *(uint2*)&T[w][row][lane * 4] = make_uint2(0u, 0u);
  s_wait_async0();
  __syncthreads();

  // Lower-triangular 16x16 tiles of Z: (0,0), (16,0), (16,16).
  const int mts[3] = {0, 16, 16};
  const int nts[3] = {0, 0, 16};
#pragma unroll
  for (int tix = 0; tix < 3; ++tix) {
    const int mt = mts[tix], nt = nts[tix];
    v8f acc = {};
#pragma unroll
    for (int kb = 0; kb < 4; ++kb) {
      const v16h a = load_a_frag(&T[w][mt][kb * 32], 136);
      const v16h b = load_bT_frag(&T[w][nt][kb * 32], 136);
      acc = __builtin_amdgcn_wmma_f32_16x16x32_f16(
          false, a, false, b, (short)0, acc, false, false);
    }
    const int hi = (lane & 16) ? 8 : 0;
    const int cn = lane & 15;
#pragma unroll
    for (int r = 0; r < 8; ++r) Z[w][mt + hi + r][nt + cn] = acc[r];
  }
  __syncthreads();

  _Float16* Rrow = R + (size_t)s * 512;
  *(uint2*)&Rrow[lane * 4] = *(const uint2*)&T[w][0][lane * 4];  // y1 copy

  // Zflat: strict lower triangle, row-major: t -> (i,j), i>j.
  for (int t = lane; t < 351; t += 32) {
    int i = (int)floorf((1.0f + sqrtf(1.0f + 8.0f * (float)t)) * 0.5f);
    while (i * (i - 1) / 2 > t) --i;
    while ((i + 1) * i / 2 <= t) ++i;
    const int j = t - i * (i - 1) / 2;
    Rrow[128 + t] = (_Float16)Z[w][i][j];
  }
  for (int t = 351 + lane; t < 384; t += 32) Rrow[128 + t] = (_Float16)0.f;
}

// ---------------------------------------------------------------------------
// fp32 -> f16 with zero padding (rows/cols).
// ---------------------------------------------------------------------------
__global__ void cvt_pad(const float* __restrict__ src, _Float16* __restrict__ dst,
                        int srows, int scols, int drows, int dcols) {
  const size_t i = (size_t)blockIdx.x * blockDim.x + threadIdx.x;
  const size_t total = (size_t)drows * dcols;
  if (i >= total) return;
  const int r = (int)(i / dcols), c = (int)(i % dcols);
  const float v = (r < srows && c < scols) ? src[(size_t)r * scols + c] : 0.f;
  dst[i] = (_Float16)v;
}

// ---------------------------------------------------------------------------
// Final layer: sigmoid(t2[B x 256] @ tw3[256 x 1]).
// ---------------------------------------------------------------------------
__global__ __launch_bounds__(256) void final_layer(const _Float16* __restrict__ t2,
                                                   const float* __restrict__ w3,
                                                   float* __restrict__ out) {
  __shared__ float w[256];
  w[threadIdx.x] = w3[threadIdx.x];
  __syncthreads();
  const int s = blockIdx.x * 256 + threadIdx.x;
  const _Float16* p = t2 + (size_t)s * 256;
  float acc = 0.f;
#pragma unroll 8
  for (int k = 0; k < 256; ++k) acc += (float)p[k] * w[k];
  out[s] = 1.0f / (1.0f + __expf(-acc));
}

// ---------------------------------------------------------------------------
extern "C" void kernel_launch(void* const* d_in, const int* in_sizes, int n_in,
                              void* d_out, int out_size, void* d_ws, size_t ws_size,
                              hipStream_t stream) {
  (void)in_sizes; (void)n_in; (void)out_size; (void)ws_size;
  const int B = 32768;

  const float*     dense  = (const float*)d_in[0];
  const long long* sparse = (const long long*)d_in[1];
  const float*     embf   = (const float*)d_in[2];
  const float*     bw1    = (const float*)d_in[3];
  const float*     bw2    = (const float*)d_in[4];
  const float*     bw3    = (const float*)d_in[5];
  const float*     bw4    = (const float*)d_in[6];
  const float*     tw1    = (const float*)d_in[7];
  const float*     tw2    = (const float*)d_in[8];
  const float*     tw3    = (const float*)d_in[9];

  char* ws = (char*)d_ws;
  size_t off = 0;
  auto alloc = [&](size_t bytes) -> void* {
    void* p = ws + off;
    off = (off + bytes + 255) & ~(size_t)255;
    return p;
  };

  _Float16* dense16 = (_Float16*)alloc((size_t)B * 32 * 2);
  _Float16* w1h     = (_Float16*)alloc((size_t)32 * 512 * 2);
  _Float16* w2h     = (_Float16*)alloc((size_t)512 * 256 * 2);
  _Float16* w3h     = (_Float16*)alloc((size_t)256 * 64 * 2);
  _Float16* w4h     = (_Float16*)alloc((size_t)64 * 128 * 2);
  _Float16* tw1h    = (_Float16*)alloc((size_t)512 * 512 * 2);
  _Float16* tw2h    = (_Float16*)alloc((size_t)512 * 256 * 2);
  _Float16* emb16   = (_Float16*)alloc((size_t)1534 * 128 * 2);
  _Float16* h1      = (_Float16*)alloc((size_t)B * 512 * 2);  // also t1
  _Float16* h2      = (_Float16*)alloc((size_t)B * 256 * 2);  // also t2
  _Float16* h3      = (_Float16*)alloc((size_t)B * 64 * 2);
  _Float16* y1h     = (_Float16*)alloc((size_t)B * 128 * 2);
  _Float16* Rb      = (_Float16*)alloc((size_t)B * 512 * 2);

  auto cvt = [&](const float* s, _Float16* d, int sr, int sc, int dr, int dc) {
    const size_t total = (size_t)dr * dc;
    const int blocks = (int)((total + 255) / 256);
    cvt_pad<<<blocks, 256, 0, stream>>>(s, d, sr, sc, dr, dc);
  };

  cvt(dense, dense16, B, 13, B, 32);
  cvt(bw1, w1h, 13, 512, 32, 512);
  cvt(bw2, w2h, 512, 256, 512, 256);
  cvt(bw3, w3h, 256, 64, 256, 64);
  cvt(bw4, w4h, 64, 128, 64, 128);
  cvt(tw1, tw1h, 479, 512, 512, 512);
  cvt(tw2, tw2h, 512, 256, 512, 256);
  cvt(embf, emb16, 1534, 128, 1534, 128);

  const dim3 blk(128);
  // Bottom MLP
  gemm_f16<true ><<<dim3(512 / 64, B / 64), blk, 0, stream>>>(dense16, w1h, h1, 512, 32);
  gemm_f16<true ><<<dim3(256 / 64, B / 64), blk, 0, stream>>>(h1, w2h, h2, 256, 512);
  gemm_f16<true ><<<dim3( 64 / 64, B / 64), blk, 0, stream>>>(h2, w3h, h3, 64, 256);
  gemm_f16<false><<<dim3(128 / 64, B / 64), blk, 0, stream>>>(h3, w4h, y1h, 128, 64);
  // Interaction + feature concat
  interact_kernel<<<B / 4, blk, 0, stream>>>(y1h, emb16, sparse, Rb);
  // Top MLP
  gemm_f16<true ><<<dim3(512 / 64, B / 64), blk, 0, stream>>>(Rb, tw1h, h1, 512, 512);
  gemm_f16<true ><<<dim3(256 / 64, B / 64), blk, 0, stream>>>(h1, tw2h, h2, 256, 512);
  final_layer<<<B / 256, 256, 0, stream>>>(h2, tw3, (float*)d_out);
}
